// CapsuleNet_29360396436003
// MI455X (gfx1250) — compile-verified, matrix operand order checked by reading, not run on hardware
//
#include <hip/hip_runtime.h>
#include <hip/hip_bf16.h>
#include <math.h>

// ---------------- problem constants ----------------
#define BB      8
#define NN      1024
#define NFEAT   128
#define NCAPS   8
#define NHID    32
#define REP     256      // NCAPS*NHID
#define MNB     16
#define ROUTIT  6
#define ROWS    8192     // BB*NN

typedef __attribute__((ext_vector_type(16))) _Float16 v16h;
typedef __attribute__((ext_vector_type(8)))  _Float16 v8h;
typedef __attribute__((ext_vector_type(8)))  float    v8f;
typedef __attribute__((ext_vector_type(4)))  float    v4f;

// ---------------- WMMA helpers ----------------
__device__ __forceinline__ v8f wmma_f16(v16h a, v16h b, v8f c) {
    // v_wmma_f32_16x16x32_f16
    return __builtin_amdgcn_wmma_f32_16x16x32_f16(false, a, false, b, (short)0, c, false, false);
}

// A-style fragment (16 rows x K=32) from row-major f16 [.. x ld].
// Per ISA 7.12.2 the 16 halves of a lane are two contiguous 8-half runs:
//   k in [h*8, h*8+8) and [16+h*8, 16+h*8+8), h = lane>>4  -> two b128 loads.
// Used for A operands AND for B operands stored transposed (row = output col).
__device__ __forceinline__ v16h loadA16(const _Float16* __restrict__ p, int ld, int row0, int k0) {
    int lane = threadIdx.x & 31;
    int r = row0 + (lane & 15);
    int h = (lane >> 4) << 3;                 // 0 or 8
    const _Float16* base = p + (size_t)r * ld + k0 + h;
    v8h lo = *(const v8h*)(base);
    v8h hi = *(const v8h*)(base + 16);
    return __builtin_shufflevector(lo, hi, 0, 1, 2, 3, 4, 5, 6, 7,
                                           8, 9, 10, 11, 12, 13, 14, 15);
}

// Same pattern from a row-major f32 source: four b128 loads + pack-convert.
__device__ __forceinline__ v16h loadA32(const float* __restrict__ p, int ld, int row0, int k0) {
    int lane = threadIdx.x & 31;
    int r = row0 + (lane & 15);
    int h = (lane >> 4) << 3;
    const float* base = p + (size_t)r * ld + k0 + h;
    v4f a0 = *(const v4f*)(base);
    v4f a1 = *(const v4f*)(base + 4);
    v4f b0 = *(const v4f*)(base + 16);
    v4f b1 = *(const v4f*)(base + 20);
    v16h a;
#pragma unroll
    for (int e = 0; e < 4; ++e) {
        a[e]      = (_Float16)a0[e];
        a[e + 4]  = (_Float16)a1[e];
        a[e + 8]  = (_Float16)b0[e];
        a[e + 12] = (_Float16)b1[e];
    }
    return a;
}

__device__ __forceinline__ float waveReduceSum(float v) {
#pragma unroll
    for (int off = 16; off > 0; off >>= 1) v += __shfl_xor(v, off, 32);
    return v;
}

// ---------------- kernel 1: PCA GEMM + leaky_relu ----------------
// H[8192,256] = leaky_relu(input[8192,128] @ pca_w[128,256] + pca_b)
// B operand consumed as pre-transposed f16 PWT[256,128].
__global__ __launch_bounds__(256) void k_pca(const float* __restrict__ A,
                                             const _Float16* __restrict__ WT,
                                             const float* __restrict__ bias,
                                             float* __restrict__ H) {
    int wave = threadIdx.x >> 5;
    int tile = blockIdx.x * 8 + wave;        // 8192 tiles: 512 x 16
    int tm = tile >> 4, tn = tile & 15;
    int row0 = tm * 16, col0 = tn * 16;
    v8f acc = {};
#pragma unroll
    for (int k0 = 0; k0 < NFEAT; k0 += 32) {
        v16h a = loadA32(A, NFEAT, row0, k0);
        v16h b = loadA16(WT, NFEAT, col0, k0);   // B^T, A-style
        acc = wmma_f16(a, b, acc);
    }
    int lane = threadIdx.x & 31;
    int col = col0 + (lane & 15);
    float bv = bias[col];
#pragma unroll
    for (int v = 0; v < 8; ++v) {
        int row = row0 + v + ((lane >> 4) << 3);
        float val = acc[v] + bv;
        H[(size_t)row * REP + col] = val > 0.f ? val : 0.01f * val;
    }
}

// ---------------- kernel 2: capsule routing layer ----------------
__global__ __launch_bounds__(256) void k_routing(const float* __restrict__ hin,
                                                 const int* __restrict__ nb,
                                                 float* __restrict__ hout) {
    __shared__ float xs[REP];
    __shared__ float us[REP];
    __shared__ float zs[MNB * REP];
    __shared__ float ps[MNB * NCAPS];
    __shared__ int   nbid[MNB];

    int t = threadIdx.x;
    int node = blockIdx.x;           // b*N + n
    int b = node >> 10;
    int n = node & (NN - 1);
    int k = t >> 5;                  // capsule for this thread (wave id)

    if (t < MNB) nbid[t] = nb[n * MNB + t];
    __syncthreads();

    {
        float v = hin[(size_t)node * REP + t];
        float ss = waveReduceSum(v * v);
        float dn = fmaxf(sqrtf(ss), 1e-12f);
        float xv = v / dn;
        xs[t] = xv;
        us[t] = xv;
    }
    for (int j = 0; j < MNB; ++j) {
        float v = hin[((size_t)b * NN + nbid[j]) * REP + t];
        float ss = waveReduceSum(v * v);
        float dn = fmaxf(sqrtf(ss), 1e-12f);
        zs[j * REP + t] = v / dn;
    }
    __syncthreads();

    for (int it = 0; it < ROUTIT; ++it) {
        if (it == 0) {
            if (t < MNB * NCAPS) ps[t] = 0.125f;     // softmax of zeros
        } else if (t < MNB * NCAPS) {
            int m = t >> 3, kk = t & 7;
            const float* zrow = &zs[m * REP + kk * NHID];
            const float* urow = &us[kk * NHID];
            float acc = 0.f;
#pragma unroll
            for (int i = 0; i < NHID; ++i) acc += zrow[i] * urow[i];
            float mx = acc;
#pragma unroll
            for (int off = 4; off > 0; off >>= 1) mx = fmaxf(mx, __shfl_xor(mx, off, 32));
            float e = __expf(acc - mx);
            float sm = e;
#pragma unroll
            for (int off = 4; off > 0; off >>= 1) sm += __shfl_xor(sm, off, 32);
            ps[t] = e / sm;
        }
        __syncthreads();
        float nu = xs[t];
#pragma unroll
        for (int m = 0; m < MNB; ++m) nu += zs[m * REP + t] * ps[m * NCAPS + k];
        if (it < ROUTIT - 1) {
            float ss = waveReduceSum(nu * nu);
            nu /= fmaxf(sqrtf(ss), 1e-12f);
        }
        us[t] = nu;
        __syncthreads();
    }
    hout[(size_t)node * REP + t] = us[t];
}

// ---------------- kernel 3a: transpose+convert attention weights ----------------
// out[which][cap][c*32 + k] = src[cap][k*32 + c]   (f32 -> f16, 32x32 transpose)
__global__ __launch_bounds__(256) void k_wcvt(const float* __restrict__ wq,
                                              const float* __restrict__ wk,
                                              const float* __restrict__ wv,
                                              const float* __restrict__ ws,
                                              _Float16* __restrict__ out) {
    int idx = blockIdx.x * 256 + threadIdx.x;    // 4 * 8192
    int which = idx >> 13, off = idx & 8191;
    int m = off >> 10;                 // capsule
    int c = (off >> 5) & 31;           // output row  (= B column)
    int kk = off & 31;                 // output col  (= B row / K)
    const float* src = which == 0 ? wq : which == 1 ? wk : which == 2 ? wv : ws;
    out[idx] = (_Float16)src[m * 1024 + kk * 32 + c];
}

// ---------------- kernel 3b: transpose+convert pca_w -> PWT[256,128] f16 ----------------
__global__ __launch_bounds__(256) void k_pwt(const float* __restrict__ W,
                                             _Float16* __restrict__ WT) {
    int idx = blockIdx.x * 256 + threadIdx.x;    // 32768
    int c = idx >> 7;                  // 0..255 output row (= B column)
    int kk = idx & 127;                // 0..127 K
    WT[idx] = (_Float16)W[kk * REP + c];
}

// ---------------- kernel 4: extract capsule slice + row sq-norms ----------------
__global__ __launch_bounds__(256) void k_extract(const float* __restrict__ H, int c,
                                                 _Float16* __restrict__ X,
                                                 float* __restrict__ sq) {
    int t = threadIdx.x;
    int row = blockIdx.x * 8 + (t >> 5);
    int d = t & 31;
    float v = H[(size_t)row * REP + c * NHID + d];
    X[(size_t)row * NHID + d] = (_Float16)v;
    float ss = waveReduceSum(v * v);
    if (d == 0) sq[row] = ss;
}

// ---------------- kernel 5: Gram via WMMA -> pairwise distances ----------------
__global__ __launch_bounds__(256) void k_gram(const _Float16* __restrict__ X,
                                              const float* __restrict__ sq,
                                              float* __restrict__ D) {
    int wave = threadIdx.x >> 5;
    int tile = blockIdx.x * 8 + wave;            // 8 * 64 * 64 tiles
    int b = tile >> 12;
    int tid = tile & 4095;
    int i0 = (tid >> 6) * 16, j0 = (tid & 63) * 16;
    const _Float16* Xb = X + (size_t)b * NN * NHID;
    v16h a  = loadA16(Xb, NHID, i0, 0);
    v16h bf = loadA16(Xb, NHID, j0, 0);          // B = X^T -> A-style load
    v8f acc = {};
    acc = wmma_f16(a, bf, acc);
    int lane = threadIdx.x & 31;
    int j = j0 + (lane & 15);
    float sqj = sq[b * NN + j];
#pragma unroll
    for (int v = 0; v < 8; ++v) {
        int i = i0 + v + ((lane >> 4) << 3);
        float d2 = fmaxf(sq[b * NN + i] + sqj - 2.f * acc[v], 0.f);
        float dd = sqrtf(d2);
        if (i == j) dd = 0.f;
        D[((size_t)b * NN + i) * NN + j] = dd;
    }
}

// ---------------- kernel 6: kth-smallest distance per row (KNN=8) ----------------
__global__ __launch_bounds__(256) void k_knn(const float* __restrict__ D,
                                             float* __restrict__ thresh) {
    __shared__ float vals[NN];
    __shared__ float rv[256];
    __shared__ int   ri[256];
    __shared__ float found;
    int t = threadIdx.x;
    const float* Dr = D + (size_t)blockIdx.x * NN;
    for (int i = t; i < NN; i += 256) vals[i] = Dr[i];
    __syncthreads();
    float th = 0.f;
    for (int s = 0; s < 8; ++s) {
        float bv = 3.4e38f; int bi = 0;
        for (int i = t; i < NN; i += 256) {
            float v = vals[i];
            if (v < bv) { bv = v; bi = i; }
        }
        rv[t] = bv; ri[t] = bi;
        __syncthreads();
        for (int off = 128; off > 0; off >>= 1) {
            if (t < off && rv[t + off] < rv[t]) { rv[t] = rv[t + off]; ri[t] = ri[t + off]; }
            __syncthreads();
        }
        if (t == 0) { found = rv[0]; vals[ri[0]] = 3.4e38f; }
        __syncthreads();
        th = found;
        __syncthreads();
    }
    if (t == 0) thresh[blockIdx.x] = th;
}

// ---------------- kernel 7: q/k/v projections via WMMA (K=32) ----------------
// Q, K stored row-major [8192,32]; V stored TRANSPOSED per batch: VT[b][c][n].
__global__ __launch_bounds__(256) void k_qkv(const _Float16* __restrict__ X,
                                             const _Float16* __restrict__ wqt,
                                             const _Float16* __restrict__ wkt,
                                             const _Float16* __restrict__ wvt,
                                             const float* __restrict__ bq,
                                             const float* __restrict__ bk,
                                             const float* __restrict__ bv,
                                             _Float16* __restrict__ Q,
                                             _Float16* __restrict__ K,
                                             _Float16* __restrict__ VT) {
    int which = blockIdx.y;
    const _Float16* WT = which == 0 ? wqt : which == 1 ? wkt : wvt;
    const float*    Bs = which == 0 ? bq : which == 1 ? bk : bv;
    int wave = threadIdx.x >> 5;
    int tile = blockIdx.x * 8 + wave;        // 512 x 2 tiles
    int row0 = (tile >> 1) * 16, col0 = (tile & 1) * 16;
    v16h a = loadA16(X, NHID, row0, 0);
    v16h b = loadA16(WT, NHID, col0, 0);     // weights pre-transposed
    v8f acc = {};
    acc = wmma_f16(a, b, acc);
    int lane = threadIdx.x & 31;
    int col = col0 + (lane & 15);
    float bvv = Bs[col];
#pragma unroll
    for (int v = 0; v < 8; ++v) {
        int row = row0 + v + ((lane >> 4) << 3);
        _Float16 r = (_Float16)(acc[v] + bvv);
        if (which == 2) {
            int bb = row >> 10, n = row & (NN - 1);
            VT[((size_t)bb * NHID + col) * NN + n] = r;   // V^T layout
        } else {
            _Float16* O = which == 0 ? Q : K;
            O[(size_t)row * NHID + col] = r;
        }
    }
}

// ---------------- kernel 8: q.k^T scores + mutual-kNN mask (in-place over D) ----------------
__global__ __launch_bounds__(256) void k_scores(const _Float16* __restrict__ Q,
                                                const _Float16* __restrict__ K,
                                                const float* __restrict__ thresh,
                                                float* __restrict__ S) {
    int wave = threadIdx.x >> 5;
    int tile = blockIdx.x * 8 + wave;
    int b = tile >> 12;
    int tid = tile & 4095;
    int i0 = (tid >> 6) * 16, j0 = (tid & 63) * 16;
    const _Float16* Qb = Q + (size_t)b * NN * NHID;
    const _Float16* Kb = K + (size_t)b * NN * NHID;
    v16h a  = loadA16(Qb, NHID, i0, 0);
    v16h bf = loadA16(Kb, NHID, j0, 0);
    v8f acc = {};
    acc = wmma_f16(a, bf, acc);
    const float scale = 0.17677669529663687f;  // 1/sqrt(32)
    int lane = threadIdx.x & 31;
    int j = j0 + (lane & 15);
    float tj = thresh[b * NN + j];
#pragma unroll
    for (int v = 0; v < 8; ++v) {
        int i = i0 + v + ((lane >> 4) << 3);
        size_t idx = ((size_t)b * NN + i) * NN + j;
        float Dv = S[idx];
        float ti = thresh[b * NN + i];
        bool msk = (Dv <= ti) && (Dv <= tj) && (i != j);
        S[idx] = msk ? acc[v] * scale : -3.4028235e38f;
    }
}

// ---------------- kernel 9: row softmax -> f16 attention ----------------
__global__ __launch_bounds__(256) void k_softmax(const float* __restrict__ S,
                                                 _Float16* __restrict__ ATT) {
    __shared__ float red[256];
    int t = threadIdx.x;
    const float* Sr = S + (size_t)blockIdx.x * NN;
    _Float16* Ar = ATT + (size_t)blockIdx.x * NN;
    float mx = -3.4028235e38f;
    for (int i = t; i < NN; i += 256) mx = fmaxf(mx, Sr[i]);
    red[t] = mx; __syncthreads();
    for (int off = 128; off > 0; off >>= 1) {
        if (t < off) red[t] = fmaxf(red[t], red[t + off]);
        __syncthreads();
    }
    float smax = red[0];
    __syncthreads();
    float e[4]; float sum = 0.f;
#pragma unroll
    for (int ii = 0; ii < 4; ++ii) {
        float sv = Sr[t + ii * 256];
        float ev = (sv > -1e37f) ? __expf(sv - smax) : 0.f;
        e[ii] = ev; sum += ev;
    }
    red[t] = sum; __syncthreads();
    for (int off = 128; off > 0; off >>= 1) {
        if (t < off) red[t] += red[t + off];
        __syncthreads();
    }
    float inv = 1.f / fmaxf(red[0], 1e-16f);
#pragma unroll
    for (int ii = 0; ii < 4; ++ii) Ar[t + ii * 256] = (_Float16)(e[ii] * inv);
}

// ---------------- kernel 10: out = att@v + x@ws + bs (WMMA, K-loop 1024) ----------------
__global__ __launch_bounds__(256) void k_attnv(const _Float16* __restrict__ ATT,
                                               const _Float16* __restrict__ VT,
                                               const _Float16* __restrict__ X,
                                               const _Float16* __restrict__ wst,
                                               const float* __restrict__ bs,
                                               float* __restrict__ O) {
    int wave = threadIdx.x >> 5;
    int tile = blockIdx.x * 8 + wave;        // 8 * 64 * 2
    int b = tile >> 7;
    int rest = tile & 127;
    int row0 = (rest >> 1) * 16, col0 = (rest & 1) * 16;
    const _Float16* Ab  = ATT + (size_t)b * NN * NN;
    const _Float16* VTb = VT + (size_t)b * NHID * NN;
    const _Float16* Xb  = X + (size_t)b * NN * NHID;
    // skip connection: x @ ws
    v16h xa = loadA16(Xb, NHID, row0, 0);
    v16h wb = loadA16(wst, NHID, col0, 0);       // ws pre-transposed
    v8f acc = {};
    acc = wmma_f16(xa, wb, acc);
    // att @ v : both operands contiguous thanks to V^T layout
    for (int k0 = 0; k0 < NN; k0 += 32) {
        v16h a  = loadA16(Ab, NN, row0, k0);
        v16h bf = loadA16(VTb, NN, col0, k0);
        acc = wmma_f16(a, bf, acc);
    }
    int lane = threadIdx.x & 31;
    int col = col0 + (lane & 15);
    float bvv = bs[col];
#pragma unroll
    for (int v = 0; v < 8; ++v) {
        int row = row0 + v + ((lane >> 4) << 3);
        O[((size_t)b * NN + row) * NHID + col] = acc[v] + bvv;
    }
}

// ---------------- kernel 11: global mean pool ----------------
__global__ __launch_bounds__(256) void k_pool(const float* __restrict__ O,
                                              float* __restrict__ out, int c) {
    __shared__ float red[256];
    int t = threadIdx.x;
    int b = blockIdx.x;
    int d = t & 31, chunk = t >> 5;
    float s = 0.f;
    for (int r = chunk * 128; r < chunk * 128 + 128; ++r)
        s += O[((size_t)b * NN + r) * NHID + d];
    red[t] = s; __syncthreads();
    if (t < 32) {
        float tot = 0.f;
#pragma unroll
        for (int w = 0; w < 8; ++w) tot += red[w * 32 + t];
        out[b * REP + c * NHID + t] = tot * (1.f / 1024.f);
    }
}

// ---------------- workspace layout ----------------
constexpr size_t SZ_H   = (size_t)ROWS * REP * 4;        // 8 MB
constexpr size_t SZ_D   = (size_t)BB * NN * NN * 4;      // 33.5 MB
constexpr size_t SZ_ATT = (size_t)BB * NN * NN * 2;      // 16.8 MB
constexpr size_t SZ_X   = (size_t)ROWS * NHID * 2;       // 512 KB
constexpr size_t SZ_SQ  = (size_t)ROWS * 4;              // 32 KB
constexpr size_t SZ_O   = (size_t)ROWS * NHID * 4;       // 1 MB
constexpr size_t SZ_W   = 4ull * NCAPS * NHID * NHID * 2;  // 64 KB
constexpr size_t SZ_PWT = (size_t)REP * NFEAT * 2;       // 64 KB

constexpr size_t OFF_H0  = 0;
constexpr size_t OFF_H1  = OFF_H0 + SZ_H;
constexpr size_t OFF_D   = OFF_H1 + SZ_H;
constexpr size_t OFF_ATT = OFF_D + SZ_D;
constexpr size_t OFF_X   = OFF_ATT + SZ_ATT;
constexpr size_t OFF_SQ  = OFF_X + SZ_X;
constexpr size_t OFF_TH  = OFF_SQ + SZ_SQ;
constexpr size_t OFF_Q   = OFF_TH + SZ_SQ;
constexpr size_t OFF_K   = OFF_Q + SZ_X;
constexpr size_t OFF_V   = OFF_K + SZ_X;
constexpr size_t OFF_O   = OFF_V + SZ_X;
constexpr size_t OFF_W   = OFF_O + SZ_O;
constexpr size_t OFF_PWT = OFF_W + SZ_W;

extern "C" void kernel_launch(void* const* d_in, const int* in_sizes, int n_in,
                              void* d_out, int out_size, void* d_ws, size_t ws_size,
                              hipStream_t stream) {
    (void)in_sizes; (void)n_in; (void)out_size; (void)ws_size;
    const float* input_ = (const float*)d_in[0];
    const int*   nb     = (const int*)d_in[1];
    // d_in[2] edge_time_ori is unused by the reference
    const float* pca_w  = (const float*)d_in[3];
    const float* pca_b  = (const float*)d_in[4];
    const float* wq = (const float*)d_in[5];
    const float* bq = (const float*)d_in[6];
    const float* wk = (const float*)d_in[7];
    const float* bk = (const float*)d_in[8];
    const float* wv = (const float*)d_in[9];
    const float* bv = (const float*)d_in[10];
    const float* wsW = (const float*)d_in[11];
    const float* bs  = (const float*)d_in[12];
    float* out = (float*)d_out;
    char* ws = (char*)d_ws;

    float*     H0   = (float*)(ws + OFF_H0);
    float*     H1   = (float*)(ws + OFF_H1);
    float*     Dbuf = (float*)(ws + OFF_D);
    _Float16*  ATT  = (_Float16*)(ws + OFF_ATT);
    _Float16*  Xc   = (_Float16*)(ws + OFF_X);
    float*     sq   = (float*)(ws + OFF_SQ);
    float*     th   = (float*)(ws + OFF_TH);
    _Float16*  Qh   = (_Float16*)(ws + OFF_Q);
    _Float16*  Kh   = (_Float16*)(ws + OFF_K);
    _Float16*  VTh  = (_Float16*)(ws + OFF_V);
    float*     Obuf = (float*)(ws + OFF_O);
    _Float16*  W16  = (_Float16*)(ws + OFF_W);
    _Float16*  WQ16 = W16;
    _Float16*  WK16 = W16 + 8192;
    _Float16*  WV16 = W16 + 16384;
    _Float16*  WS16 = W16 + 24576;
    _Float16*  PWT  = (_Float16*)(ws + OFF_PWT);

    // pre-transpose / convert weights (tiny, one-shot)
    k_pwt<<<128, 256, 0, stream>>>(pca_w, PWT);
    k_wcvt<<<128, 256, 0, stream>>>(wq, wk, wv, wsW, W16);

    // PCA + leaky_relu (WMMA)
    k_pca<<<1024, 256, 0, stream>>>(input_, PWT, pca_b, H0);
    // two routing layers (ping-pong)
    k_routing<<<ROWS, 256, 0, stream>>>(H0, nb, H1);
    k_routing<<<ROWS, 256, 0, stream>>>(H1, nb, H0);

    for (int c = 0; c < NCAPS; ++c) {
        k_extract<<<1024, 256, 0, stream>>>(H0, c, Xc, sq);
        k_gram<<<4096, 256, 0, stream>>>(Xc, sq, Dbuf);
        k_knn<<<ROWS, 256, 0, stream>>>(Dbuf, th);
        dim3 gq(128, 3);
        k_qkv<<<gq, 256, 0, stream>>>(Xc,
                                      WQ16 + c * 1024, WK16 + c * 1024, WV16 + c * 1024,
                                      bq + c * NHID, bk + c * NHID, bv + c * NHID,
                                      Qh, Kh, VTh);
        k_scores<<<4096, 256, 0, stream>>>(Qh, Kh, th, Dbuf);
        k_softmax<<<ROWS, 256, 0, stream>>>(Dbuf, ATT);
        k_attnv<<<128, 256, 0, stream>>>(ATT, VTh, Xc, WS16 + c * 1024,
                                         bs + c * NHID, Obuf);
        k_pool<<<BB, 256, 0, stream>>>(Obuf, out, c);
    }
}
